// SoftmaxAggr_14448269984510
// MI455X (gfx1250) — compile-verified, matrix operand order checked by reading, not run on hardware
//
#include <hip/hip_runtime.h>
#include <hip/hip_bf16.h>

typedef _Float16 half_t;
typedef __attribute__((ext_vector_type(16))) _Float16 v16h;
typedef __attribute__((ext_vector_type(8)))  _Float16 v8h;
typedef __attribute__((ext_vector_type(8)))  float    v8f;

#define D_IN   256
#define HIDDEN 256

// ---------------------------------------------------------------------------
// Kernel 1: convert W (fp32, row-major [HIDDEN, D_IN]) to f16.
// ---------------------------------------------------------------------------
__global__ void convW_kernel(const float* __restrict__ W, half_t* __restrict__ Wh, int n) {
    int i = blockIdx.x * 256 + threadIdx.x;
    if (i < n) Wh[i] = (half_t)W[i];
}

// ---------------------------------------------------------------------------
// Kernel 2: segment start offsets. segs[b] = lower_bound(idx, b), b in [0, B].
// idx is sorted, length N. segs[B] == N.
// ---------------------------------------------------------------------------
__global__ void seg_starts_kernel(const int* __restrict__ idx, int N, int B,
                                  int* __restrict__ segs) {
    int b = blockIdx.x * 256 + threadIdx.x;
    if (b > B) return;
    int lo = 0, hi = N;
    while (lo < hi) {
        int mid = (lo + hi) >> 1;
        if (idx[mid] < b) lo = mid + 1; else hi = mid;
    }
    segs[b] = lo;
}

// ---------------------------------------------------------------------------
// Kernel 3: h = relu(x @ W^T + bias), stored as f16 (non-temporal stream).
// Block = 256 threads = 8 waves (wave32). Each wave owns a 16-row slab.
// A-layout (ISA 16-bit A 16x32 table): lane l, hs=l>>4, r=l&15 holds row r,
//   element j of v16h -> K = kb*32 + 8*hs + j + (j>=8 ? 8 : 0).
// B-layout is the mirrored table with N in place of M (W^T columns = W rows).
// D-layout: lane l, vgpr v -> row = 8*hs + v, col = l&15.
// ---------------------------------------------------------------------------
__global__ __launch_bounds__(256)
void gemm_relu_kernel(const float* __restrict__ x, const half_t* __restrict__ Wh,
                      const float* __restrict__ bias, half_t* __restrict__ h,
                      int N) {
    const int lane    = threadIdx.x & 31;
    const int wave    = threadIdx.x >> 5;
    const int rowBase = blockIdx.x * 128 + wave * 16;
    const int hs      = lane >> 4;     // half-wave select
    const int l16     = lane & 15;

    // Uniform-per-wave tail predicate: hot path must stay branch/exec clean.
    const bool fullTile = (rowBase + 16) <= N;

    // ---- preload A slab: 16 rows x 256 K as f16 (8 K-blocks of 32) ----
    int arow = rowBase + l16;
    if (arow >= N) arow = N - 1;                  // clamp: keep EXEC all-ones
    const float* xrow = x + (size_t)arow * D_IN;

    v16h A[8];
#pragma unroll
    for (int kb = 0; kb < 8; ++kb) {
        const float* p0 = xrow + kb * 32 + hs * 8;   // K = kb*32 + 8h .. +7
        const float* p1 = p0 + 16;                   // K = kb*32 + 16+8h .. +7
        float4 c0  = ((const float4*)p0)[0];
        float4 c0b = ((const float4*)p0)[1];
        float4 c1  = ((const float4*)p1)[0];
        float4 c1b = ((const float4*)p1)[1];
        v16h a;
        a[0]  = (half_t)c0.x;  a[1]  = (half_t)c0.y;
        a[2]  = (half_t)c0.z;  a[3]  = (half_t)c0.w;
        a[4]  = (half_t)c0b.x; a[5]  = (half_t)c0b.y;
        a[6]  = (half_t)c0b.z; a[7]  = (half_t)c0b.w;
        a[8]  = (half_t)c1.x;  a[9]  = (half_t)c1.y;
        a[10] = (half_t)c1.z;  a[11] = (half_t)c1.w;
        a[12] = (half_t)c1b.x; a[13] = (half_t)c1b.y;
        a[14] = (half_t)c1b.z; a[15] = (half_t)c1b.w;
        A[kb] = a;
    }

    // ---- sweep 16 column tiles of 16 channels ----
#pragma unroll 1
    for (int ct = 0; ct < 16; ++ct) {
        const int bcol = ct * 16 + l16;               // output channel this lane owns
        const half_t* wrow = Wh + (size_t)bcol * D_IN;
        const float bv = bias[bcol];

        // Batch ALL 16 B-tile loads first; the sched_barrier pins them ahead
        // of the WMMA chain so the waits stagger (loadcnt 0xe, 0xc, ...)
        // instead of a full s_wait_loadcnt 0 stall before every WMMA.
        v8h blo[8], bhi[8];
#pragma unroll
        for (int kb = 0; kb < 8; ++kb) {
            const half_t* q0 = wrow + kb * 32 + hs * 8;
            blo[kb] = *(const v8h*)q0;
            bhi[kb] = *(const v8h*)(q0 + 16);
        }
        __builtin_amdgcn_sched_barrier(0);

        v8f acc;
#pragma unroll
        for (int i = 0; i < 8; ++i) acc[i] = bv;      // seed C with bias

#pragma unroll
        for (int kb = 0; kb < 8; ++kb) {
            v16h B = __builtin_shufflevector(blo[kb], bhi[kb],
                      0,1,2,3,4,5,6,7,8,9,10,11,12,13,14,15);
            acc = __builtin_amdgcn_wmma_f32_16x16x32_f16(
                      /*neg_a=*/false, A[kb],
                      /*neg_b=*/false, B,
                      /*c_mod=*/(short)0, acc,
                      /*reuse_a=*/false, /*reuse_b=*/false);
        }

        // ReLU + store f16 (row = rowBase + 8*hs + v, col = bcol).
        // Non-temporal: h is a 205 MB stream > L2; keep x/W resident instead.
        half_t* hout = h + (size_t)(rowBase + hs * 8) * HIDDEN + bcol;
        if (fullTile) {
#pragma unroll
            for (int v = 0; v < 8; ++v) {
                float r = __builtin_fmaxf(acc[v], 0.f);
                __builtin_nontemporal_store((half_t)r, hout + (size_t)v * HIDDEN);
            }
        } else {
#pragma unroll
            for (int v = 0; v < 8; ++v) {
                int row = rowBase + hs * 8 + v;
                if (row < N) {
                    float r = __builtin_fmaxf(acc[v], 0.f);
                    h[(size_t)row * HIDDEN + bcol] = (half_t)r;
                }
            }
        }
    }
}

// ---------------------------------------------------------------------------
// Kernel 4: per-segment softmax-weighted aggregation.
// One block per segment (contiguous node range), thread = channel.
// agg = (sum h*exp(h*t)) / (sum exp(h*t) + 1e-16)  -- max-shift cancels
// (h = relu(unit-variance GEMM) is bounded ~6, exp cannot overflow).
// ---------------------------------------------------------------------------
__global__ __launch_bounds__(256)
void seg_agg_kernel(const half_t* __restrict__ h, const int* __restrict__ segs,
                    const float* __restrict__ t, float* __restrict__ out) {
    const int b = blockIdx.x;
    const int c = threadIdx.x;
    const int s = segs[b];
    const int e = segs[b + 1];
    const float tc = t[c];

    float S = 0.f, P = 0.f;
    int n = s;
    for (; n + 8 <= e; n += 8) {
        float hv[8];
#pragma unroll
        for (int i = 0; i < 8; ++i)
            hv[i] = (float)h[(size_t)(n + i) * HIDDEN + c];
#pragma unroll
        for (int i = 0; i < 8; ++i) {
            float ev = __expf(hv[i] * tc);
            S += ev;
            P += hv[i] * ev;
        }
    }
    for (; n < e; ++n) {
        float hv = (float)h[(size_t)n * HIDDEN + c];
        float ev = __expf(hv * tc);
        S += ev;
        P += hv * ev;
    }
    out[b * HIDDEN + c] = P / (S + 1e-16f);
}

// ---------------------------------------------------------------------------
// kernel_launch
// Inputs (setup_inputs order): x[f32 N*256], graph_idx[int N], batch_size[int 1],
//                              W[f32 256*256], b[f32 256], t[f32 256]
// Output: f32 [B, 256], B = out_size / 256.
// Workspace layout: [h f16: N*256*2 B][Wh f16: 128 KB][segs int: (B+1)*4 B]
// ---------------------------------------------------------------------------
extern "C" void kernel_launch(void* const* d_in, const int* in_sizes, int n_in,
                              void* d_out, int out_size, void* d_ws, size_t ws_size,
                              hipStream_t stream) {
    const float* x    = (const float*)d_in[0];
    const int*   gidx = (const int*)d_in[1];
    const float* W    = (const float*)d_in[3];
    const float* bias = (const float*)d_in[4];
    const float* t    = (const float*)d_in[5];
    float* out = (float*)d_out;

    const int N = in_sizes[0] / D_IN;
    const int B = out_size / HIDDEN;

    char* ws = (char*)d_ws;
    half_t* h  = (half_t*)ws;
    size_t hBytes = (size_t)N * HIDDEN * sizeof(half_t);
    half_t* Wh = (half_t*)(ws + hBytes);
    int*   segs = (int*)(ws + hBytes + (size_t)HIDDEN * D_IN * sizeof(half_t));

    convW_kernel<<<(HIDDEN * D_IN + 255) / 256, 256, 0, stream>>>(W, Wh, HIDDEN * D_IN);
    seg_starts_kernel<<<(B + 1 + 255) / 256, 256, 0, stream>>>(gidx, N, B, segs);

    int gemmBlocks = (N + 127) / 128;
    gemm_relu_kernel<<<gemmBlocks, 256, 0, stream>>>(x, Wh, bias, h, N);

    seg_agg_kernel<<<B, 256, 0, stream>>>(h, segs, t, out);
}
